// HQQLinearKernel_38534446579913
// MI455X (gfx1250) — matrix operand order repaired and checked by reading, not compiled, not run on hardware
//
#include <hip/hip_runtime.h>

#define IN_F   4096
#define OUT_F  11008
#define TOKENS 2048
#define GROUPS_PER_ROW (OUT_F / 64)   // 172 groups per K-row

typedef _Float16 f16;
typedef __attribute__((ext_vector_type(16))) _Float16 v16h;
typedef __attribute__((ext_vector_type(8)))  _Float16 v8h;
typedef __attribute__((ext_vector_type(2)))  _Float16 v2h;
typedef __attribute__((ext_vector_type(8)))  float    v8f;
typedef __attribute__((ext_vector_type(4)))  float    v4f;
typedef __attribute__((ext_vector_type(4)))  int      v4i;

typedef __attribute__((address_space(1))) v4i g_v4i;   // global 16B vector
typedef __attribute__((address_space(3))) v4i l_v4i;   // LDS 16B vector

#define MT 128   // M tile per block
#define NT 128   // N tile per block
#define KT 32    // K step (one f16 WMMA K)
#define LDA 40   // f16 elems per A-tile row (32 + 8 pad -> 80B stride, bank-friendly)
#define LDB 40   // f16 elems per B-tile row (transposed: [n][k])

#if defined(__has_builtin)
#  if __has_builtin(__builtin_amdgcn_global_load_async_to_lds_b128)
#    define HAVE_ASYNC_LDS 1
#  endif
#endif

__global__ __launch_bounds__(256)
void HQQLinearKernel_wmma_f16(const float* __restrict__ x,
                              const int*   __restrict__ Wq,
                              const float* __restrict__ scale,
                              const float* __restrict__ zero,
                              const float* __restrict__ bias,
                              float*       __restrict__ out)
{
    __shared__ f16 At[MT * LDA];       // x tile,  [m][k] f16
    __shared__ f16 Bt[NT * LDB];       // W tile,  [n][k] f16 (transposed: K contiguous)
    __shared__ int Sq[2 * KT * NT];    // double-buffered staged W_q (int32 per packed byte)

    const int t    = threadIdx.x;
    const int lane = t & 31;
    const int wave = t >> 5;
    const int wm   = wave & 3;     // wave m-group: m offset = 32*wm
    const int wn   = wave >> 2;    // wave n-group: n offset = 64*wn
    const int half = lane >> 4;    // lane 0-15 vs 16-31
    const int l16  = lane & 15;

    const int n0 = blockIdx.x * NT;
    const int m0 = blockIdx.y * MT;

    // staging/dequant assignment: thread -> (k-pair 2p,2p+1 ; 8 consecutive n)
    const int p    = t >> 4;           // 0..15 -> k rows 2p, 2p+1
    const int nloc = (t & 15) << 3;    // 0,8,...,120
    // A-tile assignment: thread -> (row, 16 consecutive k)
    const int ar = t >> 1;             // 0..127
    const int ac = (t & 1) << 4;       // 0 or 16

    // ---- async stage of one W_q tile (2 rows x 8 int32 per thread) ----
    auto stage = [&](int k0s, int buf) {
        int* sbuf = &Sq[buf * (KT * NT)];
#pragma unroll
        for (int r = 0; r < 2; ++r) {
            const int kk   = k0s + 2 * p + r;
            const int krow = kk & (IN_F / 2 - 1);       // byte-row = kk % 2048
            const int* gq  = Wq + (size_t)krow * OUT_F + (n0 + nloc);
#if defined(HAVE_ASYNC_LDS)
            g_v4i* gsrc = (g_v4i*)gq;
            l_v4i* ldst = (l_v4i*)&sbuf[(2 * p + r) * NT + nloc];
            // INST_OFFSET applies to both global and LDS addresses (ISA 08 §4.4)
            __builtin_amdgcn_global_load_async_to_lds_b128(gsrc, ldst, 0,  0);
            __builtin_amdgcn_global_load_async_to_lds_b128(gsrc, ldst, 16, 0);
#else
            v4i*       dv = (v4i*)&sbuf[(2 * p + r) * NT + nloc];
            const v4i* sv = (const v4i*)gq;
            dv[0] = sv[0];
            dv[1] = sv[1];
#endif
        }
    };

    v8f acc[2][4] = {};

    stage(0, 0);                       // prologue: tile 0 in flight
    int buf = 0;

    for (int k0 = 0; k0 < IN_F; k0 += KT) {
        __syncthreads();   // protect LDS tiles from previous iteration's readers

        // issue next tile's async fetch (wrap-around dummy on last iter keeps
        // outstanding count uniform so the wait immediate is constant)
        stage((k0 + KT) & (IN_F - 1), buf ^ 1);

        // ---- A tile: fp32 global loads -> f16 -> LDS (overlaps async) ----
        const float* ga = x + (size_t)(m0 + ar) * IN_F + (k0 + ac);
        v4f f0 = *(const v4f*)(ga + 0);
        v4f f1 = *(const v4f*)(ga + 4);
        v4f f2 = *(const v4f*)(ga + 8);
        v4f f3 = *(const v4f*)(ga + 12);
        if (k0 + KT < IN_F) __builtin_prefetch(ga + KT, 0, 1);
        v8h h0, h1;
#pragma unroll
        for (int e = 0; e < 4; ++e) {
            h0[e]     = (f16)f0[e];
            h0[e + 4] = (f16)f1[e];
            h1[e]     = (f16)f2[e];
            h1[e + 4] = (f16)f3[e];
        }
        *(v8h*)&At[ar * LDA + ac]     = h0;
        *(v8h*)&At[ar * LDA + ac + 8] = h1;

        // per-(row,thread) group params (8 consecutive n never cross a 64-group)
        const int   kkp  = k0 + 2 * p;
        const int   shift = (kkp < IN_F / 2) ? 4 : 0;   // same for both rows of pair
        const int   gcol = (n0 + nloc) >> 6;
        const float sc0 = scale[(kkp + 0) * GROUPS_PER_ROW + gcol];
        const float zp0 = zero [(kkp + 0) * GROUPS_PER_ROW + gcol];
        const float sc1 = scale[(kkp + 1) * GROUPS_PER_ROW + gcol];
        const float zp1 = zero [(kkp + 1) * GROUPS_PER_ROW + gcol];

#if defined(HAVE_ASYNC_LDS)
        // 8 outstanding: 4 current-buffer (older) + 4 next-buffer. Async loads
        // complete in order per wave, so <=4 retires the current buffer.
#  if __has_builtin(__builtin_amdgcn_s_wait_asynccnt)
        __builtin_amdgcn_s_wait_asynccnt(4);
#  else
        asm volatile("s_wait_asynccnt 4" ::: "memory");
#  endif
#endif

        // ---- dequant: 2 k-rows x 8 n -> packed dword stores into Bt[n][k] ----
        const int* sbuf = &Sq[buf * (KT * NT)];
        const int* s0 = &sbuf[(2 * p + 0) * NT + nloc];
        const int* s1 = &sbuf[(2 * p + 1) * NT + nloc];
        v4i a0 = *(const v4i*)(s0);
        v4i a1 = *(const v4i*)(s0 + 4);
        v4i b0 = *(const v4i*)(s1);
        v4i b1 = *(const v4i*)(s1 + 4);
#pragma unroll
        for (int j = 0; j < 8; ++j) {
            const int q0 = (((j < 4) ? a0[j] : a1[j - 4]) >> shift) & 0xF;
            const int q1 = (((j < 4) ? b0[j] : b1[j - 4]) >> shift) & 0xF;
            v2h w;
            w[0] = (f16)(((float)q0 - zp0) * sc0);
            w[1] = (f16)(((float)q1 - zp1) * sc1);
            *(v2h*)&Bt[(nloc + j) * LDB + 2 * p] = w;   // k=2p (lo), k=2p+1 (hi)
        }

        __syncthreads();

        // ---- fragments (layouts per CDNA5 ISA §7.12.2) ----
        // A: lane m = l16; elems e<8 -> K=8*half+e ; e>=8 -> K=16+8*half+(e-8)
        v16h afrag[2];
#pragma unroll
        for (int i = 0; i < 2; ++i) {
            const int row = wm * 32 + i * 16 + l16;
            v8h lo = *(const v8h*)&At[row * LDA + 8 * half];
            v8h hi = *(const v8h*)&At[row * LDA + 16 + 8 * half];
#pragma unroll
            for (int e = 0; e < 8; ++e) { afrag[i][e] = lo[e]; afrag[i][e + 8] = hi[e]; }
        }
        // B: lane n = l16; elems e<8 -> K=8*half+e ; e>=8 -> K=16+8*half+(e-8)  (K-contiguous in Bt)
        v16h bfrag[4];
#pragma unroll
        for (int j = 0; j < 4; ++j) {
            const int row = wn * 64 + j * 16 + l16;
            v8h lo = *(const v8h*)&Bt[row * LDB + 8 * half];
            v8h hi = *(const v8h*)&Bt[row * LDB + 16 + 8 * half];
#pragma unroll
            for (int e = 0; e < 8; ++e) { bfrag[j][e] = lo[e]; bfrag[j][e + 8] = hi[e]; }
        }

#pragma unroll
        for (int i = 0; i < 2; ++i)
#pragma unroll
            for (int j = 0; j < 4; ++j)
                acc[i][j] = __builtin_amdgcn_wmma_f32_16x16x32_f16(
                    false, afrag[i], false, bfrag[j], (short)0, acc[i][j], false, false);

        buf ^= 1;
    }

#if defined(HAVE_ASYNC_LDS)
    // drain the dummy wrap-around issue before the wave ends
#  if __has_builtin(__builtin_amdgcn_s_wait_asynccnt)
    __builtin_amdgcn_s_wait_asynccnt(0);
#  else
    asm volatile("s_wait_asynccnt 0" ::: "memory");
#  endif
#endif

    // ---- epilogue: C/D layout -> M = 8*half + vgpr, N = l16 ----
#pragma unroll
    for (int j = 0; j < 4; ++j) {
        const int   n  = n0 + wn * 64 + j * 16 + l16;
        const float bv = bias[n];
#pragma unroll
        for (int i = 0; i < 2; ++i) {
            const int mb = m0 + wm * 32 + i * 16 + half * 8;
#pragma unroll
            for (int v = 0; v < 8; ++v)
                out[(size_t)(mb + v) * OUT_F + n] = acc[i][j][v] + bv;
        }
    }
}

extern "C" void kernel_launch(void* const* d_in, const int* in_sizes, int n_in,
                              void* d_out, int out_size, void* d_ws, size_t ws_size,
                              hipStream_t stream) {
    const float* x     = (const float*)d_in[0];
    const int*   Wq    = (const int*)d_in[1];   // one packed byte per int32
    const float* scale = (const float*)d_in[2];
    const float* zero  = (const float*)d_in[3];
    const float* bias  = (const float*)d_in[4];
    float*       out   = (float*)d_out;

    dim3 grid(OUT_F / NT, TOKENS / MT, 1);   // 86 x 16
    dim3 block(256, 1, 1);                   // 8 waves (wave32)
    hipLaunchKernelGGL(HQQLinearKernel_wmma_f16, grid, block, 0, stream,
                       x, Wq, scale, zero, bias, out);
}
